// LazyDecoderBlock_28424093565390
// MI455X (gfx1250) — matrix operand order repaired and checked by reading, not compiled, not run on hardware
//
#include <hip/hip_runtime.h>

// ---------------------------------------------------------------------------
// Types / helpers
// ---------------------------------------------------------------------------
typedef __bf16 v16bf __attribute__((ext_vector_type(16)));
typedef float  v8f   __attribute__((ext_vector_type(8)));
typedef unsigned int v4u __attribute__((ext_vector_type(4)));
typedef int v4i __attribute__((ext_vector_type(4)));
typedef int v8i __attribute__((ext_vector_type(8)));
typedef unsigned short ushort_t;
typedef unsigned int   uint_t;

union BFrag { uint4 u[2]; v16bf v; };

__device__ inline ushort_t f2bf(float f) {
    uint_t u = __float_as_uint(f);
    u += 0x7fffu + ((u >> 16) & 1u);   // round-to-nearest-even
    return (ushort_t)(u >> 16);
}
__device__ inline float bf2f(ushort_t h) {
    return __uint_as_float(((uint_t)h) << 16);
}

#define NEGMASK (-10000.0f)

// Problem constants
#define BN   4
#define TQ   1024
#define TK   1024
#define DIM  1024
#define NH   16
#define DH   64
#define FF   4096
#define NE   8
#define BT   (BN * TQ)          // 4096 tokens
#define PADROWS 4608            // 4096 + 8*64 worst-case padded MoE rows

// ---------------------------------------------------------------------------
// Tensor Data Mover: DMA one 2D bf16 tile (tile_d0 x tile_d1, row stride
// stride0 elems) from global into LDS at lds_off, inserting 16 bytes of pad
// after every 64 bytes (matches ushort_t[rows][40] padded LDS layout).
// D# built per CDNA5 ISA ch.8 (group0 + group1, groups 2/3 zero => 2D).
// ---------------------------------------------------------------------------
__device__ inline void tdm_load_tile_bf16(unsigned lds_off,
                                          unsigned long long gaddr,
                                          int tile_d0, int tile_d1,
                                          int stride0) {
    v4u g0;
    g0[0] = 1u;                                              // count=1 (valid user D#)
    g0[1] = lds_off;                                         // lds_addr (bytes)
    g0[2] = (unsigned)(gaddr & 0xFFFFFFFFull);               // global_addr[31:0]
    g0[3] = (unsigned)((gaddr >> 32) & 0x1FFFFFFull)         // global_addr[56:32]
          | (2u << 30);                                      // type = 2 ("image")
    v8i g1;
    g1[0] = (int)((1u << 16)      // data_size = 1 -> 2 bytes
                | (1u << 20)      // pad_enable
                | (3u << 22)      // pad_interval: 16 DWORDs (64B)
                | (3u << 25));    // pad_amount:   4 DWORDs (16B)
    unsigned td0 = (unsigned)tile_d0;      // tensor_dim0 == tile (exact fit)
    unsigned td1 = (unsigned)tile_d1;      // tensor_dim1 == tile rows
    g1[1] = (int)((td0 & 0xFFFFu) << 16);                    // [63:48]: dim0 lo16
    g1[2] = (int)((td0 >> 16) | ((td1 & 0xFFFFu) << 16));    // dim0 hi16 | dim1 lo16
    g1[3] = (int)((td1 >> 16) | ((unsigned)tile_d0 << 16));  // dim1 hi16 | tile_dim0
    g1[4] = (int)((unsigned)tile_d1);                        // tile_dim1 (tile_dim2=0)
    g1[5] = (int)(unsigned)stride0;                          // tensor_dim0_stride lo32
    g1[6] = 0;                                               // stride0 hi16 | stride1 lo16
    g1[7] = 0;                                               // stride1 hi32
    v4i z = {0, 0, 0, 0};
#if defined(__clang_major__) && (__clang_major__ >= 23)
    v8i z8 = {0, 0, 0, 0, 0, 0, 0, 0};
    __builtin_amdgcn_tensor_load_to_lds(g0, g1, z, z, z8, 0);
#else
    __builtin_amdgcn_tensor_load_to_lds(g0, g1, z, z, 0);
#endif
}

// ---------------------------------------------------------------------------
// fp32 -> bf16 transposed weight conversion:  dst[n*K + k] = bf16(src[k*N + n])
// blockIdx.z = batch (expert)
// ---------------------------------------------------------------------------
__global__ __launch_bounds__(256) void k_transpose(const float* __restrict__ src,
                                                   ushort_t* __restrict__ dst,
                                                   int K, int N) {
    size_t boff = (size_t)blockIdx.z * (size_t)K * (size_t)N;
    size_t idx  = (size_t)blockIdx.x * 256 + threadIdx.x;
    size_t tot  = (size_t)K * (size_t)N;
    if (idx >= tot) return;
    int n = (int)(idx % (size_t)N);
    int k = (int)(idx / (size_t)N);
    dst[boff + (size_t)n * K + k] = f2bf(src[boff + idx]);
}

// ---------------------------------------------------------------------------
// RMSNorm: one token per block, D=1024, output bf16
// ---------------------------------------------------------------------------
__global__ __launch_bounds__(256) void k_rms(const float* __restrict__ x,
                                             const float* __restrict__ w,
                                             ushort_t* __restrict__ out) {
    __shared__ float red[256];
    const int tid = threadIdx.x;
    const size_t t = blockIdx.x;
    const float* xr = x + t * DIM;
    float xv[4]; float ss = 0.f;
#pragma unroll
    for (int i = 0; i < 4; ++i) { xv[i] = xr[tid + 256 * i]; ss += xv[i] * xv[i]; }
    red[tid] = ss;
    __syncthreads();
    for (int s = 128; s > 0; s >>= 1) {
        if (tid < s) red[tid] += red[tid + s];
        __syncthreads();
    }
    float rs = rsqrtf(red[0] * (1.0f / DIM) + 1e-6f);
#pragma unroll
    for (int i = 0; i < 4; ++i) {
        int c = tid + 256 * i;
        out[t * DIM + c] = f2bf(xv[i] * rs * w[c]);
    }
}

// ---------------------------------------------------------------------------
// Tiled bf16 WMMA GEMM:  C[M x N] = A[M x K] * B[K x N]
// B supplied pre-transposed (Bt[N x K]) so B-fragments are K-contiguous.
// Block tile 64x128, 8 waves (4 row-waves x 2 col-waves), K-step 32.
// B tiles are staged by the Tensor Data Mover (double-buffered, TENSORcnt);
// A tiles use vector loads (MoE-1 gathers rows via perm -> not TDM-able).
// MODE 0: out bf16
// MODE 1: out fp32 = acc + Resid
// MODE 2: MoE-1  (A rows gathered via perm, expert weight/bias via bases,
//                 SiLU, out bf16)
// MODE 3: MoE-2  (expert weight/bias via bases, out fp32 scattered to token:
//                 out[tok] = Resid[tok] + score[tok]*(acc + bias))
// ---------------------------------------------------------------------------
template <int MODE>
__global__ __launch_bounds__(256) void k_gemm(const ushort_t* __restrict__ A,
                                              const ushort_t* __restrict__ Bt,
                                              void* __restrict__ Out,
                                              const float* __restrict__ Resid,
                                              const float* __restrict__ bias,
                                              const int* __restrict__ perm,
                                              const int* __restrict__ bases,
                                              const float* __restrict__ score,
                                              int M, int N, int K) {
    __shared__ ushort_t As[64][40];
    __shared__ ushort_t Bs[2][128][40];
    __shared__ int rowsrc[64];

    const int tid = threadIdx.x;
    const int rowbase = blockIdx.y * 64;
    const int nbase   = blockIdx.x * 128;

    // Expert selection for MoE tiles
    if (MODE >= 2) {
        if (rowbase >= bases[NE]) return;
        int e = 0;
        while (rowbase >= bases[e + 1]) ++e;
        Bt   += (size_t)e * (size_t)N * (size_t)K;
        bias += (size_t)e * (size_t)N;
    }
    if (MODE == 2) {
        if (tid < 64) {
            int p = perm[rowbase + tid];
            rowsrc[tid] = (p < 0) ? 0 : p;
        }
    }

    const int l  = tid & 31;
    const int wid = tid >> 5;
    const int wm = wid & 3;          // row wave 0..3
    const int wn = wid >> 2;         // col wave 0..1
    const int lr = l & 15;
    const bool hi = l >= 16;
    const int kb   = hi ? 8 : 0;     // unused; A-frag also uses kb16
    const int kb16 = hi ? 16 : 0;    // B-frag K sub-base

    v8f acc[4];
#pragma unroll
    for (int c = 0; c < 4; ++c)
#pragma unroll
        for (int i = 0; i < 8; ++i) acc[c][i] = 0.f;

    // TDM double-buffer prologue: wave 0 kicks off the first B tile.
    const unsigned long long bTileBase =
        (unsigned long long)(size_t)(Bt + (size_t)nbase * K);
    const unsigned ldsB[2] = { (unsigned)(size_t)&Bs[0][0][0],
                               (unsigned)(size_t)&Bs[1][0][0] };
    if (wid == 0) {
        tdm_load_tile_bf16(ldsB[0], bTileBase, 32, 128, K);
    }

    for (int k0 = 0; k0 < K; k0 += 32) {
        const int buf = (k0 >> 5) & 1;
        __syncthreads();                       // As reusable, prev compute done
        {   // A tile: 64x32, 8 elems/thread (vector path; supports gather)
            int r = tid >> 2, c = (tid & 3) * 8;
            int src = (MODE == 2) ? rowsrc[r] : (rowbase + r);
            const ushort_t* ap = A + (size_t)src * K + k0 + c;
            *(uint4*)&As[r][c] = *(const uint4*)ap;
            if (k0 + 32 < K) __builtin_prefetch(ap + 32, 0, 3);
        }
        if (wid == 0) {
            if (k0 + 32 < K) {
                // async-launch next B tile into the other buffer
                tdm_load_tile_bf16(ldsB[buf ^ 1],
                                   bTileBase + (size_t)(k0 + 32) * 2, 32, 128, K);
                __builtin_amdgcn_s_wait_tensorcnt(1);   // current buf complete
            } else {
                __builtin_amdgcn_s_wait_tensorcnt(0);
            }
        }
        __syncthreads();                       // As + Bs[buf] visible to all

        BFrag a;
        a.u[0] = *(const uint4*)&As[16 * wm + lr][kb16];
        a.u[1] = *(const uint4*)&As[16 * wm + lr][kb16 + 8];
#pragma unroll
        for (int c = 0; c < 4; ++c) {
            BFrag b;
            const ushort_t* bp = &Bs[buf][64 * wn + 16 * c + lr][kb16];
            b.u[0] = *(const uint4*)bp;
            b.u[1] = *(const uint4*)(bp + 8);
            acc[c] = __builtin_amdgcn_wmma_f32_16x16x32_bf16(
                false, a.v, false, b.v, (short)0, acc[c], false, false);
        }
    }

    // Epilogue
#pragma unroll
    for (int c = 0; c < 4; ++c) {
        int gcol = nbase + 64 * wn + 16 * c + lr;
#pragma unroll
        for (int v = 0; v < 8; ++v) {
            int grow = rowbase + 16 * wm + v + (hi ? 8 : 0);
            float o = acc[c][v];
            if (MODE == 0) {
                ((ushort_t*)Out)[(size_t)grow * N + gcol] = f2bf(o);
            } else if (MODE == 1) {
                size_t idx = (size_t)grow * N + gcol;
                ((float*)Out)[idx] = o + Resid[idx];
            } else if (MODE == 2) {
                o += bias[gcol];
                o = o / (1.0f + __expf(-o));           // SiLU
                ((ushort_t*)Out)[(size_t)grow * N + gcol] = f2bf(o);
            } else {                                    // MODE == 3
                int tok = perm[grow];
                if (tok >= 0) {
                    size_t idx = (size_t)tok * N + gcol;
                    ((float*)Out)[idx] =
                        Resid[idx] + score[tok] * (o + bias[gcol]);
                }
            }
        }
    }
}

// ---------------------------------------------------------------------------
// Flash attention: 128 q-rows per block (8 waves x 16 rows), dh = 64.
// K tile staged row-major (B-frag of Q*K^T reads K rows contiguously);
// V tile staged transposed (B-frag of P*V reads contiguously);
// P converted D-layout -> A-layout through per-wave LDS.
// KVF32: K/V are fp32 (cross-attn context) converted on load.
// ---------------------------------------------------------------------------
template <bool CAUSAL, bool KVF32>
__global__ __launch_bounds__(256) void k_attn(const ushort_t* __restrict__ Q, int qRS,
                                              const void* __restrict__ Kp, int kRS,
                                              const void* __restrict__ Vp, int vRS,
                                              ushort_t* __restrict__ Y, int yRS,
                                              int Tk, int kvShift, float scale) {
    __shared__ ushort_t Ks[32][72];
    __shared__ ushort_t Vs[64][40];
    __shared__ ushort_t Ps[8][16][32];

    const int tid = threadIdx.x;
    const int b = blockIdx.y >> 4;
    const int h = blockIdx.y & 15;
    const int hk = h >> kvShift;
    const int q0 = blockIdx.x * 128;

    const int l  = tid & 31;
    const int wid = tid >> 5;
    const int lr = l & 15;
    const bool hi = l >= 16;
    const int kb   = hi ? 8 : 0;
    const int kb16 = hi ? 16 : 0;

    const ushort_t* Qb = Q + (size_t)b * TQ * qRS + h * DH;
    ushort_t*       Yb = Y + (size_t)b * TQ * yRS + h * DH;
    const size_t kBase = (size_t)b * Tk * kRS + hk * DH;
    const size_t vBase = (size_t)b * Tk * vRS + hk * DH;
    const ushort_t* Ku = (const ushort_t*)Kp;
    const float*    Kf = (const float*)Kp;
    const ushort_t* Vu = (const ushort_t*)Vp;
    const float*    Vf = (const float*)Vp;

    // Load Q fragments (stay in registers across the whole k-loop)
    BFrag qa[2];
    {
        const ushort_t* qp = Qb + (size_t)(q0 + wid * 16 + lr) * qRS;
#pragma unroll
        for (int s = 0; s < 2; ++s) {
            qa[s].u[0] = *(const uint4*)(qp + 32 * s + kb16);
            qa[s].u[1] = *(const uint4*)(qp + 32 * s + kb16 + 8);
        }
    }

    float m[8], lsum[8];
    v8f O[4];
#pragma unroll
    for (int v = 0; v < 8; ++v) { m[v] = -1e30f; lsum[v] = 0.f; }
#pragma unroll
    for (int c = 0; c < 4; ++c)
#pragma unroll
        for (int i = 0; i < 8; ++i) O[c][i] = 0.f;

    int kend = Tk;
    if (CAUSAL) { int lim = q0 + 128; kend = lim < Tk ? lim : Tk; }

    for (int kt = 0; kt < kend; kt += 32) {
        __syncthreads();
        {   // stage K tile (32 keys x 64) row-major bf16
            int kr = tid >> 3, c0 = (tid & 7) * 8;
            if constexpr (KVF32) {
                const float* kp = Kf + kBase + (size_t)(kt + kr) * kRS + c0;
                alignas(16) ushort_t t8[8];
#pragma unroll
                for (int j = 0; j < 8; ++j) t8[j] = f2bf(kp[j]);
                *(uint4*)&Ks[kr][c0] = *(const uint4*)t8;
            } else {
                *(uint4*)&Ks[kr][c0] =
                    *(const uint4*)(Ku + kBase + (size_t)(kt + kr) * kRS + c0);
            }
            // stage V tile transposed: Vs[dh][key]
            if constexpr (KVF32) {
                const float* vp = Vf + vBase + (size_t)(kt + kr) * vRS + c0;
#pragma unroll
                for (int j = 0; j < 8; ++j) Vs[c0 + j][kr] = f2bf(vp[j]);
            } else {
                const ushort_t* vp = Vu + vBase + (size_t)(kt + kr) * vRS + c0;
#pragma unroll
                for (int j = 0; j < 8; ++j) Vs[c0 + j][kr] = vp[j];
            }
        }
        __syncthreads();

        // S = Q * K^T for 32 keys: two 16x16 score tiles, K-contraction 64
        v8f S0, S1;
#pragma unroll
        for (int i = 0; i < 8; ++i) { S0[i] = 0.f; S1[i] = 0.f; }
#pragma unroll
        for (int s = 0; s < 2; ++s) {
            BFrag bk;
            bk.u[0] = *(const uint4*)&Ks[lr][32 * s + kb16];
            bk.u[1] = *(const uint4*)&Ks[lr][32 * s + kb16 + 8];
            S0 = __builtin_amdgcn_wmma_f32_16x16x32_bf16(
                false, qa[s].v, false, bk.v, (short)0, S0, false, false);
            BFrag bk2;
            bk2.u[0] = *(const uint4*)&Ks[16 + lr][32 * s + kb16];
            bk2.u[1] = *(const uint4*)&Ks[16 + lr][32 * s + kb16 + 8];
            S1 = __builtin_amdgcn_wmma_f32_16x16x32_bf16(
                false, qa[s].v, false, bk2.v, (short)0, S1, false, false);
        }

        // Online softmax update, row-wise (rows live across 16-lane halves)
#pragma unroll
        for (int v = 0; v < 8; ++v) {
            int rowg = q0 + wid * 16 + v + (hi ? 8 : 0);
            float s0 = S0[v] * scale;
            float s1 = S1[v] * scale;
            if (CAUSAL) {
                if (kt + lr > rowg)      s0 += NEGMASK;
                if (kt + 16 + lr > rowg) s1 += NEGMASK;
            }
            float tm = fmaxf(s0, s1);
#pragma unroll
            for (int off = 1; off < 16; off <<= 1)
                tm = fmaxf(tm, __shfl_xor(tm, off, 16));
            float nm = fmaxf(m[v], tm);
            float al = __expf(m[v] - nm);
            m[v] = nm;
            float p0 = __expf(s0 - nm);
            float p1 = __expf(s1 - nm);
            float rs = p0 + p1;
#pragma unroll
            for (int off = 1; off < 16; off <<= 1)
                rs += __shfl_xor(rs, off, 16);
            lsum[v] = lsum[v] * al + rs;
#pragma unroll
            for (int c = 0; c < 4; ++c) O[c][v] *= al;
            int prow = v + (hi ? 8 : 0);
            Ps[wid][prow][lr]      = f2bf(p0);
            Ps[wid][prow][16 + lr] = f2bf(p1);
        }
        // intra-wave LDS RAW hazard (cross-lane): explicit DS-counter wait
        asm volatile("s_wait_dscnt 0x0" ::: "memory");

        BFrag pf;
        pf.u[0] = *(const uint4*)&Ps[wid][lr][kb16];
        pf.u[1] = *(const uint4*)&Ps[wid][lr][kb16 + 8];
#pragma unroll
        for (int c = 0; c < 4; ++c) {
            BFrag vb;
            vb.u[0] = *(const uint4*)&Vs[16 * c + lr][kb16];
            vb.u[1] = *(const uint4*)&Vs[16 * c + lr][kb16 + 8];
            O[c] = __builtin_amdgcn_wmma_f32_16x16x32_bf16(
                false, pf.v, false, vb.v, (short)0, O[c], false, false);
        }
    }

    // Write normalized output (bf16)
#pragma unroll
    for (int c = 0; c < 4; ++c) {
        int col = 16 * c + lr;
#pragma unroll
        for (int v = 0; v < 8; ++v) {
            int rowl = wid * 16 + v + (hi ? 8 : 0);
            float o = O[c][v] / lsum[v];
            Yb[(size_t)(q0 + rowl) * yRS + col] = f2bf(o);
        }
    }
}

// ---------------------------------------------------------------------------
// MoE routing kernels
// ---------------------------------------------------------------------------
__global__ __launch_bounds__(256) void k_init(int* __restrict__ perm,
                                              int* __restrict__ counts) {
    int g = blockIdx.x * 256 + threadIdx.x;
    if (g < PADROWS) perm[g] = -1;
    if (g < NE) counts[g] = 0;
}

__global__ __launch_bounds__(256) void k_gate(const ushort_t* __restrict__ xm,
                                              const float* __restrict__ wg,
                                              int* __restrict__ counts,
                                              int* __restrict__ eid,
                                              int* __restrict__ posv,
                                              float* __restrict__ scorev) {
    __shared__ float xs[DIM];
    __shared__ float dots[NE];
    const int tid = threadIdx.x;
    const size_t t = blockIdx.x;
#pragma unroll
    for (int i = 0; i < 4; ++i) {
        int c = tid + 256 * i;
        xs[c] = bf2f(xm[t * DIM + c]);
    }
    __syncthreads();
    int wid = tid >> 5, l = tid & 31;
    float s = 0.f;
    for (int d = l; d < DIM; d += 32) s += xs[d] * wg[(size_t)d * NE + wid];
#pragma unroll
    for (int off = 16; off >= 1; off >>= 1) s += __shfl_xor(s, off, 32);
    if (l == 0) dots[wid] = s;
    __syncthreads();
    if (tid == 0) {
        float mx = dots[0]; int bi = 0;
#pragma unroll
        for (int e = 1; e < NE; ++e)
            if (dots[e] > mx) { mx = dots[e]; bi = e; }
        float den = 0.f;
#pragma unroll
        for (int e = 0; e < NE; ++e) den += __expf(dots[e] - mx);
        int p = atomicAdd(&counts[bi], 1);
        eid[t] = bi; posv[t] = p; scorev[t] = 1.0f / den;
    }
}

__global__ void k_prefix(const int* __restrict__ counts, int* __restrict__ bases) {
    if (threadIdx.x == 0 && blockIdx.x == 0) {
        int acc = 0;
        bases[0] = 0;
        for (int e = 0; e < NE; ++e) {
            acc += ((counts[e] + 63) >> 6) << 6;   // pad each bucket to 64 rows
            bases[e + 1] = acc;
        }
    }
}

__global__ __launch_bounds__(256) void k_scatter(const int* __restrict__ eid,
                                                 const int* __restrict__ posv,
                                                 const int* __restrict__ bases,
                                                 int* __restrict__ perm) {
    int t = blockIdx.x * 256 + threadIdx.x;
    if (t < BT) perm[bases[eid[t]] + posv[t]] = t;
}

// ---------------------------------------------------------------------------
// Host orchestration
// ---------------------------------------------------------------------------
extern "C" void kernel_launch(void* const* d_in, const int* in_sizes, int n_in,
                              void* d_out, int out_size, void* d_ws, size_t ws_size,
                              hipStream_t stream) {
    (void)in_sizes; (void)n_in; (void)out_size; (void)ws_size;

    const float* x         = (const float*)d_in[0];
    const float* k_ctx     = (const float*)d_in[1];
    const float* v_ctx     = (const float*)d_in[2];
    const float* sa_norm_w = (const float*)d_in[3];
    const float* w_qkv     = (const float*)d_in[4];
    const float* sa_wout   = (const float*)d_in[5];
    const float* ca_norm_w = (const float*)d_in[6];
    const float* w_q       = (const float*)d_in[7];
    const float* ca_wout   = (const float*)d_in[8];
    const float* moe_norm_w= (const float*)d_in[9];
    const float* w_gate    = (const float*)d_in[10];
    const float* w1        = (const float*)d_in[11];
    const float* b1        = (const float*)d_in[12];
    const float* w2        = (const float*)d_in[13];
    const float* b2        = (const float*)d_in[14];

    char* wsb = (char*)d_ws;
    size_t off = 0;
    auto take = [&](size_t bytes) -> char* {
        char* p = wsb + off;
        off = (off + bytes + 255) & ~(size_t)255;
        return p;
    };

    ushort_t* XN    = (ushort_t*)take((size_t)BT * DIM * 2);
    ushort_t* QKV   = (ushort_t*)take((size_t)BT * 3 * DIM * 2);
    ushort_t* Yb    = (ushort_t*)take((size_t)BT * DIM * 2);
    ushort_t* QC    = (ushort_t*)take((size_t)BT * DIM * 2);
    float*    Xf    = (float*)   take((size_t)BT * DIM * 4);
    ushort_t* WQKVT = (ushort_t*)take((size_t)DIM * 3 * DIM * 2);
    ushort_t* WSOT  = (ushort_t*)take((size_t)DIM * DIM * 2);
    ushort_t* WQT   = (ushort_t*)take((size_t)DIM * DIM * 2);
    ushort_t* WCOT  = (ushort_t*)take((size_t)DIM * DIM * 2);
    ushort_t* W1T   = (ushort_t*)take((size_t)NE * DIM * FF * 2);
    ushort_t* W2T   = (ushort_t*)take((size_t)NE * FF * DIM * 2);
    ushort_t* Hbuf  = (ushort_t*)take((size_t)PADROWS * FF * 2);
    int*   COUNTS   = (int*)  take(NE * 4);
    int*   BASES    = (int*)  take((NE + 1) * 4);
    int*   EID      = (int*)  take((size_t)BT * 4);
    int*   POS      = (int*)  take((size_t)BT * 4);
    float* SCORE    = (float*)take((size_t)BT * 4);
    int*   PERM     = (int*)  take((size_t)PADROWS * 4);

    const dim3 blk(256);
    const float scale = 0.125f;   // 1/sqrt(64)

    // ---- weight conversion (fp32 -> bf16, transposed) ----
    k_transpose<<<dim3((DIM * 3 * DIM + 255) / 256, 1, 1), blk, 0, stream>>>(w_qkv, WQKVT, DIM, 3 * DIM);
    k_transpose<<<dim3((DIM * DIM + 255) / 256, 1, 1), blk, 0, stream>>>(sa_wout, WSOT, DIM, DIM);
    k_transpose<<<dim3((DIM * DIM + 255) / 256, 1, 1), blk, 0, stream>>>(w_q, WQT, DIM, DIM);
    k_transpose<<<dim3((DIM * DIM + 255) / 256, 1, 1), blk, 0, stream>>>(ca_wout, WCOT, DIM, DIM);
    k_transpose<<<dim3((DIM * FF + 255) / 256, 1, NE), blk, 0, stream>>>(w1, W1T, DIM, FF);
    k_transpose<<<dim3((FF * DIM + 255) / 256, 1, NE), blk, 0, stream>>>(w2, W2T, FF, DIM);

    // ---- self-attention ----
    k_rms<<<dim3(BT), blk, 0, stream>>>(x, sa_norm_w, XN);
    k_gemm<0><<<dim3(3 * DIM / 128, BT / 64), blk, 0, stream>>>(
        XN, WQKVT, QKV, nullptr, nullptr, nullptr, nullptr, nullptr, BT, 3 * DIM, DIM);
    k_attn<true, false><<<dim3(TQ / 128, BN * NH), blk, 0, stream>>>(
        QKV, 3 * DIM, (const void*)(QKV + DIM), 3 * DIM, (const void*)(QKV + 2 * DIM), 3 * DIM,
        Yb, DIM, TK, 0, scale);
    k_gemm<1><<<dim3(DIM / 128, BT / 64), blk, 0, stream>>>(
        Yb, WSOT, Xf, x, nullptr, nullptr, nullptr, nullptr, BT, DIM, DIM);

    // ---- cross-attention (GQA) ----
    k_rms<<<dim3(BT), blk, 0, stream>>>(Xf, ca_norm_w, XN);
    k_gemm<0><<<dim3(DIM / 128, BT / 64), blk, 0, stream>>>(
        XN, WQT, QC, nullptr, nullptr, nullptr, nullptr, nullptr, BT, DIM, DIM);
    k_attn<false, true><<<dim3(TQ / 128, BN * NH), blk, 0, stream>>>(
        QC, DIM, (const void*)k_ctx, 4 * DH, (const void*)v_ctx, 4 * DH,
        Yb, DIM, TK, 2, scale);
    k_gemm<1><<<dim3(DIM / 128, BT / 64), blk, 0, stream>>>(
        Yb, WCOT, Xf, Xf, nullptr, nullptr, nullptr, nullptr, BT, DIM, DIM);

    // ---- top-1 MoE FFN ----
    k_rms<<<dim3(BT), blk, 0, stream>>>(Xf, moe_norm_w, XN);
    k_init<<<dim3((PADROWS + 255) / 256), blk, 0, stream>>>(PERM, COUNTS);
    k_gate<<<dim3(BT), blk, 0, stream>>>(XN, w_gate, COUNTS, EID, POS, SCORE);
    k_prefix<<<dim3(1), dim3(32), 0, stream>>>(COUNTS, BASES);
    k_scatter<<<dim3(BT / 256), blk, 0, stream>>>(EID, POS, BASES, PERM);
    k_gemm<2><<<dim3(FF / 128, PADROWS / 64), blk, 0, stream>>>(
        XN, W1T, Hbuf, nullptr, b1, PERM, BASES, nullptr, PADROWS, FF, DIM);
    k_gemm<3><<<dim3(DIM / 128, PADROWS / 64), blk, 0, stream>>>(
        Hbuf, W2T, d_out, Xf, b2, PERM, BASES, SCORE, PADROWS, DIM, FF);
}
